// LLTM_84705345012109
// MI455X (gfx1250) — matrix operand order repaired and checked by reading, not compiled, not run on hardware
//
#include <hip/hip_runtime.h>
#include <hip/hip_bf16.h>
#include <stdint.h>

// LLTM fused kernel for gfx1250 (MI455X), round 2.
// gates = [old_h|input] @ W via bf16x3 split-precision WMMA
// (v_wmma_f32_16x16x32_bf16), fp32 accumulate, fused activation epilogue.
// Register-staged software pipeline overlaps global loads with WMMA.

typedef __attribute__((ext_vector_type(16))) __bf16 v16bf;
typedef __attribute__((ext_vector_type(8)))  float  v8f;

#define B_DIM 4096
#define F_DIM 2048
#define S_DIM 2048
#define K_DIM 4096   // F + S
#define N3    6144   // 3*S
#define BM    128
#define BNG   64     // per-gate N tile
#define BK    32
#define ASTR  36     // A row stride in bf16 elems (32 + 4 pad, keeps 8B align)
#define BSTR  36
#define NCHUNK (K_DIM / BK)

union Frag { uint32_t u[8]; v16bf v; };

__global__ __launch_bounds__(256)
void lltm_fused_wmma(const float* __restrict__ W,
                     const float* __restrict__ bias,
                     const float* __restrict__ input,
                     const float* __restrict__ old_h,
                     const float* __restrict__ old_c,
                     float* __restrict__ out)
{
    __shared__ alignas(16) __bf16 sAhi[BM * ASTR];
    __shared__ alignas(16) __bf16 sAlo[BM * ASTR];
    __shared__ alignas(16) __bf16 sBhi[3 * BNG * BSTR];
    __shared__ alignas(16) __bf16 sBlo[3 * BNG * BSTR];

    const int t    = threadIdx.x;
    const int lane = t & 31;
    const int wave = t >> 5;
    const int wm   = wave >> 1;   // 0..3 : 32-row slice
    const int wn   = wave & 1;    // 0..1 : 32-col slice per gate
    const int h    = lane >> 4;   // lane half select
    const int ln   = lane & 15;

    const int mBlock = blockIdx.y * BM;
    const int nBlock = blockIdx.x * BNG;   // per-gate column base (s index)

    // acc[iMsub][jNsub][gate]
    v8f acc[2][2][3];
    #pragma unroll
    for (int i = 0; i < 2; ++i)
        #pragma unroll
        for (int j = 0; j < 2; ++j)
            #pragma unroll
            for (int g = 0; g < 3; ++g) {
                v8f z = {0.f,0.f,0.f,0.f,0.f,0.f,0.f,0.f};
                acc[i][j][g] = z;
            }

    // A staging: 8 threads per 32-wide row, 32 rows/pass, 4 passes.
    const int ar  = t >> 3;        // 0..31
    const int ac4 = (t & 7) * 4;   // 0,4,...,28

    float4 gA[4];   // staged A chunk (128x32 f32 across the block)
    float4 gB[6];   // staged B chunk (32x192 f32 across the block)

    // ---- preload chunk 0 ----
    {
        const float* xsrc = old_h;             // k0 = 0 < S
        #pragma unroll
        for (int p = 0; p < 4; ++p)
            gA[p] = *(const float4*)(xsrc + (size_t)(mBlock + p * 32 + ar) * S_DIM + ac4);
        #pragma unroll
        for (int g = 0; g < 3; ++g)
            #pragma unroll
            for (int q = 0; q < 2; ++q) {
                const int flat = q * 256 + t;
                const int kr = flat >> 4;          // 0..31
                const int n4 = (flat & 15) * 4;    // 0..60
                gB[g * 2 + q] = *(const float4*)(W + (size_t)kr * N3
                                                   + (size_t)g * S_DIM + nBlock + n4);
            }
    }

    for (int chunk = 0; chunk < NCHUNK; ++chunk) {
        __syncthreads();   // previous iteration's LDS reads complete

        // ---- store staged chunk to LDS with bf16 hi/lo split ----
        #pragma unroll
        for (int p = 0; p < 4; ++p) {
            union { __bf16 b[4]; uint2 u2; } uh, ul;
            #pragma unroll
            for (int jj = 0; jj < 4; ++jj) {
                const float x = (&gA[p].x)[jj];
                const __bf16 hi = (__bf16)x;
                uh.b[jj] = hi;
                ul.b[jj] = (__bf16)(x - (float)hi);
            }
            const int r = p * 32 + ar;
            *(uint2*)(sAhi + r * ASTR + ac4) = uh.u2;
            *(uint2*)(sAlo + r * ASTR + ac4) = ul.u2;
        }
        #pragma unroll
        for (int g = 0; g < 3; ++g)
            #pragma unroll
            for (int q = 0; q < 2; ++q) {
                const int flat = q * 256 + t;
                const int kr = flat >> 4;
                const int n4 = (flat & 15) * 4;
                #pragma unroll
                for (int jj = 0; jj < 4; ++jj) {
                    const float x = (&gB[g * 2 + q].x)[jj];
                    const __bf16 hi = (__bf16)x;
                    sBhi[(g * BNG + n4 + jj) * BSTR + kr] = hi;
                    sBlo[(g * BNG + n4 + jj) * BSTR + kr] = (__bf16)(x - (float)hi);
                }
            }

        __syncthreads();   // LDS tiles ready

        // ---- stage next chunk's globals (overlaps with WMMA below) ----
        if (chunk + 1 < NCHUNK) {
            const int k0n = (chunk + 1) * BK;
            const float* xn = (k0n < S_DIM) ? old_h : input;
            const int kkn = k0n & (S_DIM - 1);
            #pragma unroll
            for (int p = 0; p < 4; ++p)
                gA[p] = *(const float4*)(xn + (size_t)(mBlock + p * 32 + ar) * S_DIM + kkn + ac4);
            #pragma unroll
            for (int g = 0; g < 3; ++g)
                #pragma unroll
                for (int q = 0; q < 2; ++q) {
                    const int flat = q * 256 + t;
                    const int kr = flat >> 4;
                    const int n4 = (flat & 15) * 4;
                    gB[g * 2 + q] = *(const float4*)(W + (size_t)(k0n + kr) * N3
                                                       + (size_t)g * S_DIM + nBlock + n4);
                }
            // gfx1250 prefetch hint for the chunk after next (W stream)
            if (chunk + 2 < NCHUNK)
                __builtin_prefetch(W + (size_t)(k0n + BK + (t >> 4)) * N3 + nBlock + (t & 15) * 4, 0, 0);
        }

        // ---- A fragments (hi & lo) for this wave's two M-subtiles ----
        Frag fahi[2], falo[2];
        #pragma unroll
        for (int i = 0; i < 2; ++i) {
            const uint32_t* arh = (const uint32_t*)(sAhi + (wm * 32 + i * 16 + ln) * ASTR);
            const uint32_t* arl = (const uint32_t*)(sAlo + (wm * 32 + i * 16 + ln) * ASTR);
            const int o = 4 * h;
            #pragma unroll
            for (int q = 0; q < 4; ++q) {
                fahi[i].u[q]     = arh[o + q];        // K = 8h + 2q, 8h + 2q + 1
                fahi[i].u[4 + q] = arh[o + 8 + q];    // K = 16 + 8h + 2q ...
                falo[i].u[q]     = arl[o + q];
                falo[i].u[4 + q] = arl[o + 8 + q];
            }
        }

        // ---- 3 gates x 2 N-subtiles x 2 M-subtiles x 3 split-products ----
        #pragma unroll
        for (int g = 0; g < 3; ++g) {
            #pragma unroll
            for (int j = 0; j < 2; ++j) {
                Frag fbh, fbl;
                const uint32_t* brh =
                    (const uint32_t*)(sBhi + (g * BNG + wn * 32 + j * 16 + ln) * BSTR);
                const uint32_t* brl =
                    (const uint32_t*)(sBlo + (g * BNG + wn * 32 + j * 16 + ln) * BSTR);
                const int ob = 4 * h;
                #pragma unroll
                for (int q = 0; q < 4; ++q) {
                    fbh.u[q] = brh[ob + q];         fbl.u[q] = brl[ob + q];
                    fbh.u[4 + q] = brh[ob + 8 + q]; fbl.u[4 + q] = brl[ob + 8 + q];
                }

                #pragma unroll
                for (int i = 0; i < 2; ++i) {
                    acc[i][j][g] = __builtin_amdgcn_wmma_f32_16x16x32_bf16(
                        false, fahi[i].v, false, fbh.v, (short)0, acc[i][j][g], false, false);
                    acc[i][j][g] = __builtin_amdgcn_wmma_f32_16x16x32_bf16(
                        false, fahi[i].v, false, fbl.v, (short)0, acc[i][j][g], false, false);
                    acc[i][j][g] = __builtin_amdgcn_wmma_f32_16x16x32_bf16(
                        false, falo[i].v, false, fbh.v, (short)0, acc[i][j][g], false, false);
                }
            }
        }
    }

    // ---- Fused epilogue: bias + sigmoid/sigmoid/ELU + cell update + tanh ----
    #pragma unroll
    for (int j = 0; j < 2; ++j) {
        const int sCol = nBlock + wn * 32 + j * 16 + ln;
        const float b_i = bias[sCol];
        const float b_o = bias[S_DIM + sCol];
        const float b_c = bias[2 * S_DIM + sCol];
        #pragma unroll
        for (int i = 0; i < 2; ++i) {
            #pragma unroll
            for (int r = 0; r < 8; ++r) {
                const int m = mBlock + wm * 32 + i * 16 + r + 8 * h;
                const float gi = acc[i][j][0][r] + b_i;
                const float go = acc[i][j][1][r] + b_o;
                const float gc = acc[i][j][2][r] + b_c;
                const float ig = 1.f / (1.f + __expf(-gi));
                const float og = 1.f / (1.f + __expf(-go));
                const float cand = (gc > 0.f) ? gc : (__expf(gc) - 1.f);
                const float ncell = old_c[(size_t)m * S_DIM + sCol] + cand * ig;
                const float e2 = __expf(-2.f * ncell);
                const float nh = ((1.f - e2) / (1.f + e2)) * og;
                out[(size_t)m * S_DIM + sCol] = nh;
                out[(size_t)B_DIM * S_DIM + (size_t)m * S_DIM + sCol] = ncell;
            }
        }
    }
}

extern "C" void kernel_launch(void* const* d_in, const int* in_sizes, int n_in,
                              void* d_out, int out_size, void* d_ws, size_t ws_size,
                              hipStream_t stream) {
    (void)in_sizes; (void)n_in; (void)out_size; (void)d_ws; (void)ws_size;
    const float* W     = (const float*)d_in[0];
    const float* bias  = (const float*)d_in[1];
    const float* input = (const float*)d_in[2];
    const float* old_h = (const float*)d_in[3];
    const float* old_c = (const float*)d_in[4];
    float* out = (float*)d_out;

    dim3 grid(S_DIM / BNG, B_DIM / BM);   // 32 x 32 = 1024 blocks
    lltm_fused_wmma<<<grid, 256, 0, stream>>>(W, bias, input, old_h, old_c, out);
}
